// CovNeXtBlock_1924145349040
// MI455X (gfx1250) — matrix operand order repaired and checked by reading, not compiled, not run on hardware
//
#include <hip/hip_runtime.h>
#include <hip/hip_bf16.h>
#include <math.h>

typedef __attribute__((ext_vector_type(16))) _Float16 v16h;
typedef __attribute__((ext_vector_type(8)))  float    v8f;

#define C_CH 96
#define H_CH 384
#define SPD  64
#define DHW  262144            // 64^3
#define ROWS 128               // spatial rows per MLP block
#define SXP  104               // sX / sO row stride in halves (16B aligned rows)
#define SHP  392               // sH row stride in halves (16B aligned rows)

union FragU { v16h v; unsigned u[8]; };

// Branch-free GELU (tanh form): exact enough here since the whole MLP branch
// is multiplied by gamma=1e-6 before the fp32 residual add. Uses raw
// v_exp_f32 / v_rcp_f32 so no libm EXEC-divergent slow paths appear between
// WMMAs.
__device__ __forceinline__ float gelu_fast(float v) {
  const float u  = 0.7978845608028654f * (v + 0.044715f * v * v * v);
  const float au = fabsf(u);
  // tanh(|u|) = (1 - e^-2|u|) / (1 + e^-2|u|);  e^-2|u| = exp2(-2|u|*log2 e)
  const float e  = __builtin_amdgcn_exp2f(-2.8853900817779268f * au);
  float th = (1.f - e) * __builtin_amdgcn_rcpf(1.f + e);
  th = copysignf(th, u);
  return 0.5f * v * (1.f + th);
}

// -------------------------------------------------------------------------
// Kernel 1: depthwise 7x7x7 conv (pad 3) + bias -> f16 y, NCDHW layout.
// Thread tile: 4 outputs along D x 2 along W (register-sliding D window).
// -------------------------------------------------------------------------
__global__ __launch_bounds__(256) void k_dwconv(
    const float* __restrict__ x, const float* __restrict__ dw_w,
    const float* __restrict__ dw_b, _Float16* __restrict__ y)
{
  const int tid = threadIdx.x;
  const int w0  = (tid & 31) * 2;            // 0..62, even
  const int h   = blockIdx.y * 8 + (tid >> 5);
  const int d0  = blockIdx.x * 4;
  const int bc  = blockIdx.z;                // b*96 + c
  const int c   = bc % C_CH;

  const float* xc = x + bc * DHW;
  const float* wc = dw_w + c * 343;

  float acc[4][2];
  #pragma unroll
  for (int i = 0; i < 4; ++i) { acc[i][0] = 0.f; acc[i][1] = 0.f; }

  for (int dz = 0; dz < 10; ++dz) {          // input planes d0-3 .. d0+6
    const int zz = d0 - 3 + dz;
    if (zz < 0 || zz >= SPD) continue;
    for (int ky = 0; ky < 7; ++ky) {
      const int hy = h - 3 + ky;
      if (hy < 0 || hy >= SPD) continue;
      const float* rowp = xc + (zz * SPD + hy) * SPD;
      float win[8];
      #pragma unroll
      for (int j = 0; j < 8; ++j) {
        const int wx = w0 - 3 + j;
        win[j] = (wx >= 0 && wx < SPD) ? rowp[wx] : 0.f;
      }
      #pragma unroll
      for (int od = 0; od < 4; ++od) {
        const int kd = dz - od;              // weight d-index = zz-(d0+od)+3
        if (kd < 0 || kd >= 7) continue;
        const float* wrow = wc + (kd * 7 + ky) * 7;
        #pragma unroll
        for (int kx = 0; kx < 7; ++kx) {
          const float wgt = wrow[kx];
          acc[od][0] = fmaf(wgt, win[kx],     acc[od][0]);
          acc[od][1] = fmaf(wgt, win[kx + 1], acc[od][1]);
        }
      }
    }
  }

  const float bias = dw_b[c];
  _Float16* yb = y + bc * DHW + h * SPD + w0;
  #pragma unroll
  for (int od = 0; od < 4; ++od) {
    union { _Float16 h2[2]; unsigned u; } pk;
    pk.h2[0] = (_Float16)(acc[od][0] + bias);
    pk.h2[1] = (_Float16)(acc[od][1] + bias);
    *(unsigned*)(yb + (d0 + od) * (SPD * SPD)) = pk.u;   // coalesced b32
  }
}

// -------------------------------------------------------------------------
// Kernel 2: LayerNorm + MLP (WMMA f16) + GELU + layer-scale + residual.
// 256 threads = 8 waves; 128 rows/block; each wave owns 16 rows.
// -------------------------------------------------------------------------
__global__ __launch_bounds__(256) void k_mlp(
    const _Float16* __restrict__ y, const float* __restrict__ x,
    const float* __restrict__ ln_g, const float* __restrict__ ln_b,
    const float* __restrict__ w1, const float* __restrict__ b1,
    const float* __restrict__ w2, const float* __restrict__ b2,
    const float* __restrict__ gamma, float* __restrict__ out)
{
  extern __shared__ char smem[];
  _Float16* sW1 = (_Float16*)smem;             // [384][96] f16, W1 transposed (n-major)
  _Float16* sW2 = sW1 + C_CH * H_CH;           // [96][384] f16, W2 transposed (n-major)
  _Float16* sX  = sW2 + H_CH * C_CH;           // [128][SXP] LN'd A-tile
  _Float16* sH  = sX  + ROWS * SXP;            // [128][SHP] hidden tile
  _Float16* sO  = sH  + ROWS * SHP;            // [128][SXP] out tile
  float*    sB1 = (float*)(sO + ROWS * SXP);   // [384]
  float*    sB2 = sB1 + H_CH;                  // [96]
  float*    sG  = sB2 + C_CH;                  // [96]
  float*    sLg = sG  + C_CH;                  // [96]
  float*    sLb = sLg + C_CH;                  // [96]

  const int t = threadIdx.x;

  // ---- stage weights/consts into LDS (f32 -> f16) ----
  for (int i = t; i < C_CH * H_CH; i += 256) sW1[(i % H_CH) * C_CH + i / H_CH] = (_Float16)w1[i];
  for (int i = t; i < H_CH * C_CH; i += 256) sW2[(i % C_CH) * H_CH + i / C_CH] = (_Float16)w2[i];
  for (int i = t; i < H_CH; i += 256) sB1[i] = b1[i];
  if (t < C_CH) { sB2[t] = b2[t]; sG[t] = gamma[t]; sLg[t] = ln_g[t]; sLb[t] = ln_b[t]; }

  // ---- load 96ch x 128row conv tile (channel-major global -> row-major LDS) ----
  const long rowBase = (long)blockIdx.x * ROWS;
  const int  batch   = (int)(rowBase >> 18);        // 2^18 = 64^3
  const int  sp      = (int)(rowBase & (DHW - 1));
  const _Float16* yb = y + (long)batch * C_CH * DHW + sp;
  for (int i = t; i < C_CH * ROWS; i += 256) {
    const int c = i >> 7, r = i & 127;
    sX[r * SXP + c] = yb[(long)c * DHW + r];        // contiguous 128-row runs
  }
  __syncthreads();

  // ---- LayerNorm over channels, in place, f32 math ----
  if (t < ROWS) {
    _Float16* rp = sX + t * SXP;
    float s = 0.f, ss = 0.f;
    #pragma unroll 8
    for (int c = 0; c < C_CH; ++c) { const float v = (float)rp[c]; s += v; ss += v * v; }
    const float mu  = s * (1.f / C_CH);
    const float var = ss * (1.f / C_CH) - mu * mu;
    const float rs  = rsqrtf(var + 1e-5f);
    #pragma unroll 8
    for (int c = 0; c < C_CH; ++c) {
      const float v = (float)rp[c];
      rp[c] = (_Float16)((v - mu) * rs * sLg[c] + sLb[c]);
    }
  }
  __syncthreads();

  const int lane  = t & 31;
  const int wv    = t >> 5;                 // wave 0..7
  const int rB    = wv * 16;                // 16 rows per wave
  const int hiK   = (lane >= 16) ? 8 : 0;   // A/C-fragment lane-half K/M offset
  const int nlane = lane & 15;

  // ---- hoist the 3 A fragments (K = 0..95) for matmul1 ----
  FragU a[3];
  {
    const _Float16* ap = sX + (rB + nlane) * SXP;
    #pragma unroll
    for (int kb = 0; kb < 3; ++kb) {
      const uint4 qa = *(const uint4*)(ap + kb * 32 + hiK);
      const uint4 qb = *(const uint4*)(ap + kb * 32 + hiK + 16);
      a[kb].u[0] = qa.x; a[kb].u[1] = qa.y; a[kb].u[2] = qa.z; a[kb].u[3] = qa.w;
      a[kb].u[4] = qb.x; a[kb].u[5] = qb.y; a[kb].u[6] = qb.z; a[kb].u[7] = qb.w;
    }
  }

  // ---- matmul1: hidden = GELU(A @ W1 + b1), 24 N-tiles x 3 K-steps ----
  for (int nt = 0; nt < 24; ++nt) {
    const int n0 = nt * 16;
    v8f acc = {0.f, 0.f, 0.f, 0.f, 0.f, 0.f, 0.f, 0.f};
    #pragma unroll
    for (int kb = 0; kb < 3; ++kb) {
      FragU bf;                                        // B: lane<->N, halves<->K
      const _Float16* bp = sW1 + (n0 + nlane) * C_CH + kb * 32 + hiK;
      const uint4 q0 = *(const uint4*)(bp);
      const uint4 q1 = *(const uint4*)(bp + 16);
      bf.u[0] = q0.x; bf.u[1] = q0.y; bf.u[2] = q0.z; bf.u[3] = q0.w;
      bf.u[4] = q1.x; bf.u[5] = q1.y; bf.u[6] = q1.z; bf.u[7] = q1.w;
      acc = __builtin_amdgcn_wmma_f32_16x16x32_f16(
              false, a[kb].v, false, bf.v, (short)0, acc, false, false);
    }
    const float bn = sB1[n0 + nlane];
    #pragma unroll
    for (int r = 0; r < 8; ++r) {
      const float v = gelu_fast(acc[r] + bn);
      sH[(rB + r + hiK) * SHP + n0 + nlane] = (_Float16)v;
    }
  }
  __syncthreads();

  // ---- matmul2: out = H @ W2 + b2 ----
  // Loop nest: K outer (hoist A fragment once per K-step), 6 N-tile
  // accumulators held live (48 VGPRs) -> 6x fewer A-side LDS loads.
  {
    v8f acc2[6];
    #pragma unroll
    for (int nt = 0; nt < 6; ++nt)
      acc2[nt] = (v8f){0.f, 0.f, 0.f, 0.f, 0.f, 0.f, 0.f, 0.f};

    for (int kb = 0; kb < 12; ++kb) {
      FragU af;
      const _Float16* ap = sH + (rB + nlane) * SHP + kb * 32 + hiK;
      const uint4 qa = *(const uint4*)(ap);
      const uint4 qb = *(const uint4*)(ap + 16);
      af.u[0] = qa.x; af.u[1] = qa.y; af.u[2] = qa.z; af.u[3] = qa.w;
      af.u[4] = qb.x; af.u[5] = qb.y; af.u[6] = qb.z; af.u[7] = qb.w;
      #pragma unroll
      for (int nt = 0; nt < 6; ++nt) {
        FragU bf;
        const _Float16* bp = sW2 + (nt * 16 + nlane) * H_CH + kb * 32 + hiK;
        const uint4 q0 = *(const uint4*)(bp);
        const uint4 q1 = *(const uint4*)(bp + 16);
        bf.u[0] = q0.x; bf.u[1] = q0.y; bf.u[2] = q0.z; bf.u[3] = q0.w;
        bf.u[4] = q1.x; bf.u[5] = q1.y; bf.u[6] = q1.z; bf.u[7] = q1.w;
        acc2[nt] = __builtin_amdgcn_wmma_f32_16x16x32_f16(
                     false, af.v, false, bf.v, (short)0, acc2[nt], false, false);
      }
    }

    #pragma unroll
    for (int nt = 0; nt < 6; ++nt) {
      const int n0 = nt * 16;
      const float bn = sB2[n0 + nlane];
      #pragma unroll
      for (int r = 0; r < 8; ++r)
        sO[(rB + r + hiK) * SXP + n0 + nlane] = (_Float16)(acc2[nt][r] + bn);
    }
  }
  __syncthreads();

  // ---- layer scale + residual, coalesced NCDHW store via LDS transpose ----
  const float* xr   = x   + (long)batch * C_CH * DHW + sp;
  float*       outr = out + (long)batch * C_CH * DHW + sp;
  for (int i = t; i < C_CH * ROWS; i += 256) {
    const int c = i >> 7, r = i & 127;
    const long g = (long)c * DHW + r;
    outr[g] = xr[g] + sG[c] * (float)sO[r * SXP + c];
  }
}

// -------------------------------------------------------------------------
extern "C" void kernel_launch(void* const* d_in, const int* in_sizes, int n_in,
                              void* d_out, int out_size, void* d_ws, size_t ws_size,
                              hipStream_t stream) {
  (void)in_sizes; (void)n_in; (void)out_size; (void)ws_size;
  const float* x    = (const float*)d_in[0];
  const float* dw_w = (const float*)d_in[1];
  const float* dw_b = (const float*)d_in[2];
  const float* ln_g = (const float*)d_in[3];
  const float* ln_b = (const float*)d_in[4];
  const float* w1   = (const float*)d_in[5];
  const float* b1   = (const float*)d_in[6];
  const float* w2   = (const float*)d_in[7];
  const float* b2   = (const float*)d_in[8];
  const float* gam  = (const float*)d_in[9];
  float* out = (float*)d_out;
  _Float16* y = (_Float16*)d_ws;              // f16 conv output, NCDHW, 100.7 MB

  // depthwise conv: 16 d-tiles x 8 h-tiles x (2 batches * 96 channels)
  k_dwconv<<<dim3(16, 8, 2 * C_CH), 256, 0, stream>>>(x, dw_w, dw_b, y);

  // LN + WMMA MLP: 524288 rows / 128 = 4096 blocks; ~304 KB dynamic LDS
  const size_t smemBytes =
      (size_t)(C_CH * H_CH + H_CH * C_CH + ROWS * SXP + ROWS * SHP + ROWS * SXP) * 2
      + (size_t)(H_CH + 4 * C_CH) * 4;
  (void)hipFuncSetAttribute((const void*)k_mlp,
                            hipFuncAttributeMaxDynamicSharedMemorySize,
                            (int)smemBytes);
  k_mlp<<<dim3((2 * DHW) / ROWS), 256, smemBytes, stream>>>(
      y, x, ln_g, ln_b, w1, b1, w2, b2, gam, out);
}
